// SpiralNet_2808908612153
// MI455X (gfx1250) — matrix operand order repaired and checked by reading, not compiled, not run on hardware
//
#include <hip/hip_runtime.h>
#include <hip/hip_bf16.h>

// ---------------------------------------------------------------------------
// SpiralNet autoencoder on MI455X (gfx1250, wave32, WMMA).
// All dense math uses v_wmma_f32_16x16x32_bf16 (bf16 operands, f32 accum).
// Weights are pre-swizzled into B-fragment order so each lane loads its whole
// 32x16 B tile with two b128 loads; gathered activations are staged in LDS and
// read as A fragments with two ds_load_b128 per lane (layouts per cdna5 ISA
// 7.12.2). The latent Wz GEMM (16 x 81920 x 256, pure weight streaming) is
// split-K across 640 waves to saturate HBM, with a deterministic fixed-order
// reduction pass.
// ---------------------------------------------------------------------------

typedef __attribute__((ext_vector_type(16))) __bf16 v16bf;
typedef __attribute__((ext_vector_type(8)))  float  v8f;

struct alignas(16) U4   { unsigned x, y, z, w; };
struct alignas(16) U4x2 { U4 lo, hi; };

__device__ __forceinline__ v16bf make_frag(U4 lo, U4 hi) {
    U4x2 t{lo, hi};
    return __builtin_bit_cast(v16bf, t);
}

__device__ __forceinline__ unsigned short f2bf(float f) {
    unsigned u = __builtin_bit_cast(unsigned, f);
    u += 0x7FFFu + ((u >> 16) & 1u);            // round-to-nearest-even
    return (unsigned short)(u >> 16);
}
__device__ __forceinline__ float bf2f(unsigned short h) {
    unsigned u = ((unsigned)h) << 16;
    return __builtin_bit_cast(float, u);
}
__device__ __forceinline__ float elu1(float x) {
    return x > 0.f ? x : (__expf(x) - 1.f);
}
__device__ __forceinline__ v8f wmma_bf16(v16bf a, v16bf b, v8f c) {
    return __builtin_amdgcn_wmma_f32_16x16x32_bf16(
        false, a, false, b, (short)0, c, false, false);
}

// ---- model constants -------------------------------------------------------
#define NB   16
#define V0N  20480
#define V1N  5120
#define V2N  1280
#define V3N  320
#define SSP  9
#define ZSPLIT 40                 // K-slices for the Wz GEMM (2560 kb / 40 = 64)

// ---------------------------------------------------------------------------
// f32 -> bf16 elementwise
__global__ void k_cvt_bf16(const float* __restrict__ in,
                           unsigned short* __restrict__ out, int n) {
    int i = blockIdx.x * 256 + threadIdx.x;
    if (i < n) out[i] = f2bf(in[i]);
}

// Swizzle a row-major f32 weight [K,N] into bf16 B-fragment blocks:
// out[(kb*NT + nt)*512 + lane*16 + e] = W[kb*32 + (e&7) + 8*(lane>>4) + 16*(e>>3)][nt*16 + (lane&15)]
// (zero padded outside K,N).
__global__ void k_swizzleB(const float* __restrict__ W,
                           unsigned short* __restrict__ out,
                           int K, int N, int NT, int total) {
    int i = blockIdx.x * 256 + threadIdx.x;
    if (i >= total) return;
    int e    = i & 15;
    int lane = (i >> 4) & 31;
    int blk  = i >> 9;
    int nt   = blk % NT;
    int kb   = blk / NT;
    int k = kb * 32 + (e & 7) + 8 * (lane >> 4) + 16 * (e >> 3);
    int n = nt * 16 + (lane & 15);
    float v = (k < K && n < N) ? W[(size_t)k * N + n] : 0.f;
    out[i] = f2bf(v);
}

// Sparse pool: out[b][vo][c] = sum_k w[vo][k] * in[b][idx[vo][k]][c], K=3.
__global__ void k_pool(const unsigned short* __restrict__ in,
                       unsigned short* __restrict__ out,
                       const int* __restrict__ idx,
                       const float* __restrict__ w,
                       int VO, int VIN, int C, int total) {
    int t = blockIdx.x * 256 + threadIdx.x;
    if (t >= total) return;
    int c  = t % C;
    int vo = (t / C) % VO;
    int b  = t / (C * VO);
    const int*   ip = idx + vo * 3;
    const float* wp = w   + vo * 3;
    float s = 0.f;
#pragma unroll
    for (int k = 0; k < 3; ++k)
        s += wp[k] * bf2f(in[((size_t)b * VIN + ip[k]) * C + c]);
    out[t] = f2bf(s);
}

// ---------------------------------------------------------------------------
// Spiral conv, tiny C_in (=3) path: K_total = 27 padded to one 32-K block.
// C_out = 64 -> 4 N-tiles, one per wave. Block = 128 threads.
__global__ __launch_bounds__(128) void k_conv0(
    const unsigned short* __restrict__ act,   // [B,V,3] bf16
    const int*            __restrict__ idx,   // [V,9]
    const unsigned short* __restrict__ wswz,  // 1 kb x 4 nt blocks
    const float*          __restrict__ bias,  // [64]
    unsigned short*       __restrict__ out,   // [B,V,64]
    int V) {
    __shared__ __align__(16) unsigned short sA[16 * 32];
    const int tid  = threadIdx.x;
    const int lane = tid & 31, wave = tid >> 5;
    const int tilesPerB = V >> 4;
    const int b     = blockIdx.x / tilesPerB;
    const int vbase = (blockIdx.x % tilesPerB) << 4;

    for (int e = tid; e < 16 * 32; e += 128) {   // uniform trip count
        int r = e >> 5, k = e & 31;
        unsigned short v = 0;
        if (k < 27) {
            int s = k / 3, c = k - s * 3;
            int nb = idx[(vbase + r) * 9 + s];
            v = act[((size_t)b * V + nb) * 3 + c];
        }
        sA[e] = v;
    }
    __syncthreads();

    const int hi = lane >> 4, row = lane & 15;
    v16bf a = make_frag(*(const U4*)(sA + row * 32 + 8 * hi),
                        *(const U4*)(sA + row * 32 + 16 + 8 * hi));
    const U4* pb = (const U4*)(wswz + (size_t)wave * 512 + lane * 16);
    v16bf bb = make_frag(pb[0], pb[1]);
    v8f acc = {};
    acc = wmma_bf16(a, bb, acc);

    int n = wave * 16 + row;
    float bv = bias[n];
#pragma unroll
    for (int r2 = 0; r2 < 8; ++r2) {
        int m = r2 + 8 * hi;
        out[((size_t)b * V + vbase + m) * 64 + n] = f2bf(elu1(acc[r2] + bv));
    }
}

// Spiral conv, general path. CIN in {64,128,256}; NTILES = C_out_pad/16.
// One workgroup (4 waves) per 16-row M tile; LDS-staged gather per spiral step.
template <int CIN, int NTILES, int NREAL, bool DOELU, bool F32OUT>
__global__ __launch_bounds__(128) void k_sconv(
    const unsigned short* __restrict__ act,   // [B,V,CIN] bf16
    const int*            __restrict__ idx,   // [V,9]
    const unsigned short* __restrict__ wswz,  // [KB][NTILES][512] bf16
    const float*          __restrict__ bias,  // [NREAL]
    void*                 __restrict__ outv,  // bf16 [B,V,NREAL] or f32
    int V) {
    constexpr int KBS = CIN / 32;             // K-blocks per spiral step
    constexpr int NTW = (NTILES + 3) / 4;     // N-tiles per wave (max)
    __shared__ __align__(16) unsigned short sA[16 * CIN];

    const int tid  = threadIdx.x;
    const int lane = tid & 31, wave = tid >> 5;
    const int tilesPerB = V >> 4;
    const int b     = blockIdx.x / tilesPerB;
    const int vbase = (blockIdx.x % tilesPerB) << 4;
    const int hi = lane >> 4, row = lane & 15;

    v8f acc[NTW] = {};

    for (int s = 0; s < SSP; ++s) {
        // stage 16 gathered rows x CIN channels (pair granularity, uniform trips)
        for (int e = tid; e < 16 * (CIN / 2); e += 128) {
            int r  = e / (CIN / 2);
            int cp = e - r * (CIN / 2);
            int nb = idx[(vbase + r) * 9 + s];
            const unsigned* src =
                (const unsigned*)(act + ((size_t)b * V + nb) * CIN);
            ((unsigned*)sA)[r * (CIN / 2) + cp] = src[cp];
        }
        __syncthreads();

#pragma unroll
        for (int kk = 0; kk < KBS; ++kk) {
            v16bf a = make_frag(
                *(const U4*)(sA + row * CIN + kk * 32 + 8 * hi),
                *(const U4*)(sA + row * CIN + kk * 32 + 16 + 8 * hi));
            int kb = s * KBS + kk;
#pragma unroll
            for (int t = 0; t < NTW; ++t) {
                int nt = wave + 4 * t;
                if (nt < NTILES) {                 // wave-uniform guard
                    const U4* pb = (const U4*)(wswz +
                        ((size_t)kb * NTILES + nt) * 512 + lane * 16);
                    v16bf bb = make_frag(pb[0], pb[1]);
                    acc[t] = wmma_bf16(a, bb, acc[t]);
                }
            }
        }
        __syncthreads();
    }

#pragma unroll
    for (int t = 0; t < NTW; ++t) {
        int nt = wave + 4 * t;
        if (nt >= NTILES) continue;
        int n = nt * 16 + row;
        float bv = (n < NREAL) ? bias[n] : 0.f;
#pragma unroll
        for (int r2 = 0; r2 < 8; ++r2) {
            int m = r2 + 8 * hi;
            float v = acc[t][r2] + bv;
            if (DOELU) v = elu1(v);
            if (n < NREAL) {
                size_t o = ((size_t)b * V + vbase + m) * NREAL + n;
                if (F32OUT) ((float*)outv)[o] = v;
                else ((unsigned short*)outv)[o] = f2bf(v);
            }
        }
    }
}

// ---------------------------------------------------------------------------
// Latent GEMM z = A[16,81920] @ Wz, split-K: blockIdx.x = N-tile (0..15),
// blockIdx.y = K-slice (0..ZSPLIT-1). Each wave streams 64 K-blocks and
// writes f32 partials; 640 concurrent waves keep HBM busy.
__global__ __launch_bounds__(32) void k_gemm_z_split(
    const unsigned short* __restrict__ A,    // [16, 81920] bf16
    const unsigned short* __restrict__ Bw,   // [2560][16][512] bf16
    float*                __restrict__ part) // [ZSPLIT][16][256] f32
{
    const int lane = threadIdx.x & 31;
    const int nt   = blockIdx.x;             // 0..15
    const int sp   = blockIdx.y;             // 0..ZSPLIT-1
    const int hi = lane >> 4, row = lane & 15;
    constexpr int KBC = (81920 / 32) / ZSPLIT;   // 64
    const int kb0 = sp * KBC;
    v8f acc = {};
    for (int kk = 0; kk < KBC; ++kk) {
        int kb = kb0 + kk;
        if (kk + 1 < KBC)
            __builtin_prefetch(Bw + ((size_t)(kb + 1) * 16 + nt) * 512 +
                               lane * 16, 0, 0);
        v16bf a = make_frag(
            *(const U4*)(A + (size_t)row * 81920 + kb * 32 + 8 * hi),
            *(const U4*)(A + (size_t)row * 81920 + kb * 32 + 16 + 8 * hi));
        const U4* pb = (const U4*)(Bw + ((size_t)kb * 16 + nt) * 512 + lane * 16);
        v16bf bb = make_frag(pb[0], pb[1]);
        acc = wmma_bf16(a, bb, acc);
    }
    int n = nt * 16 + row;
#pragma unroll
    for (int r = 0; r < 8; ++r) {
        int m = r + 8 * hi;
        part[((size_t)sp * 16 + m) * 256 + n] = acc[r];
    }
}

// Deterministic fixed-order reduction over the K-slices + bias; emits both
// the f32 z (to d_out) and the bf16 copy feeding the Wp GEMM.
__global__ void k_z_final(const float* __restrict__ part,
                          const float* __restrict__ bias,
                          float* __restrict__ zf,
                          unsigned short* __restrict__ zb) {
    int i = blockIdx.x * 256 + threadIdx.x;
    if (i >= 16 * 256) return;
    int n = i & 255;
    float s = bias[n];
    for (int sp = 0; sp < ZSPLIT; ++sp)
        s += part[(size_t)sp * 4096 + i];
    zf[i] = s;
    zb[i] = f2bf(s);
}

// Latent GEMM d = z[16,256] @ Wp + bp -> [16, 81920] bf16.
__global__ __launch_bounds__(32) void k_gemm_p(
    const unsigned short* __restrict__ A,    // [16,256] bf16
    const unsigned short* __restrict__ Bw,   // [8][5120][512] bf16
    const float*          __restrict__ bias, // [81920]
    unsigned short*       __restrict__ out)  // [16,81920] bf16
{
    const int lane = threadIdx.x & 31;
    const int nt   = blockIdx.x;             // 0..5119
    const int hi = lane >> 4, row = lane & 15;
    v8f acc = {};
#pragma unroll
    for (int kb = 0; kb < 8; ++kb) {
        v16bf a = make_frag(
            *(const U4*)(A + row * 256 + kb * 32 + 8 * hi),
            *(const U4*)(A + row * 256 + kb * 32 + 16 + 8 * hi));
        const U4* pb = (const U4*)(Bw + ((size_t)kb * 5120 + nt) * 512 + lane * 16);
        v16bf bb = make_frag(pb[0], pb[1]);
        acc = wmma_bf16(a, bb, acc);
    }
    int n = nt * 16 + row;
    float bv = bias[n];
#pragma unroll
    for (int r = 0; r < 8; ++r) {
        int m = r + 8 * hi;
        out[(size_t)m * 81920 + n] = f2bf(acc[r] + bv);
    }
}

// ---------------------------------------------------------------------------
extern "C" void kernel_launch(void* const* d_in, const int* in_sizes, int n_in,
                              void* d_out, int out_size, void* d_ws, size_t ws_size,
                              hipStream_t stream) {
    const float* x      = (const float*)d_in[0];
    const float* enc_W0 = (const float*)d_in[1];
    const float* enc_b0 = (const float*)d_in[2];
    const float* enc_W1 = (const float*)d_in[3];
    const float* enc_b1 = (const float*)d_in[4];
    const float* enc_W2 = (const float*)d_in[5];
    const float* enc_b2 = (const float*)d_in[6];
    const float* Wz     = (const float*)d_in[7];
    const float* bz     = (const float*)d_in[8];
    const float* Wp     = (const float*)d_in[9];
    const float* bp     = (const float*)d_in[10];
    const float* dec_W0 = (const float*)d_in[11];
    const float* dec_b0 = (const float*)d_in[12];
    const float* dec_W1 = (const float*)d_in[13];
    const float* dec_b1 = (const float*)d_in[14];
    const float* dec_W2 = (const float*)d_in[15];
    const float* dec_b2 = (const float*)d_in[16];
    const float* dpw0 = (const float*)d_in[17];
    const float* dpw1 = (const float*)d_in[18];
    const float* dpw2 = (const float*)d_in[19];
    const float* upw0 = (const float*)d_in[20];
    const float* upw1 = (const float*)d_in[21];
    const float* upw2 = (const float*)d_in[22];
    const int* idx0 = (const int*)d_in[23];
    const int* idx1 = (const int*)d_in[24];
    const int* idx2 = (const int*)d_in[25];
    const int* dpi0 = (const int*)d_in[26];
    const int* dpi1 = (const int*)d_in[27];
    const int* dpi2 = (const int*)d_in[28];
    const int* upi0 = (const int*)d_in[29];
    const int* upi1 = (const int*)d_in[30];
    const int* upi2 = (const int*)d_in[31];

    // ---- workspace layout (bf16 element offsets) --------------------------
    constexpr size_t W0S  = 0;                  // 1kb x 4nt
    constexpr size_t W1S  = W0S  + 2048;        // 18 x 8
    constexpr size_t W2S  = W1S  + 73728;       // 36 x 16
    constexpr size_t WZS  = W2S  + 294912;      // 2560 x 16
    constexpr size_t WPS  = WZS  + 20971520;    // 8 x 5120
    constexpr size_t WD0S = WPS  + 20971520;    // 72 x 8
    constexpr size_t WD1S = WD0S + 294912;      // 36 x 4
    constexpr size_t WD2S = WD1S + 73728;       // 18 x 1
    constexpr size_t XBF  = WD2S + 9216;
    constexpr size_t ZBF  = XBF  + 983040;
    constexpr size_t BUF0 = ZBF  + 4096;
    constexpr size_t BUF1 = BUF0 + 20971520;
    constexpr size_t ZPRT = BUF1 + 20971520;    // ZSPLIT*16*256 f32 partials

    unsigned short* ws = (unsigned short*)d_ws;
    unsigned short* xbf  = ws + XBF;
    unsigned short* zbf  = ws + ZBF;
    unsigned short* buf0 = ws + BUF0;
    unsigned short* buf1 = ws + BUF1;
    float*          zprt = (float*)(ws + ZPRT);

    auto swz = [&](const float* W, size_t off, int K, int N, int NT) {
        int KB = (K + 31) / 32;
        int total = KB * NT * 512;
        k_swizzleB<<<(total + 255) / 256, 256, 0, stream>>>(W, ws + off, K, N, NT, total);
    };
    // ---- weight swizzles --------------------------------------------------
    swz(enc_W0, W0S,  27,    64,    4);
    swz(enc_W1, W1S,  576,   128,   8);
    swz(enc_W2, W2S,  1152,  256,   16);
    swz(Wz,     WZS,  81920, 256,   16);
    swz(Wp,     WPS,  256,   81920, 5120);
    swz(dec_W0, WD0S, 2304,  128,   8);
    swz(dec_W1, WD1S, 1152,  64,    4);
    swz(dec_W2, WD2S, 576,   3,     1);

    // ---- input to bf16 ----------------------------------------------------
    {
        int n = NB * V0N * 3;
        k_cvt_bf16<<<(n + 255) / 256, 256, 0, stream>>>(x, xbf, n);
    }

    // ---- encoder ----------------------------------------------------------
    k_conv0<<<NB * V0N / 16, 128, 0, stream>>>(xbf, idx0, ws + W0S, enc_b0, buf0, V0N);
    {
        int total = NB * V1N * 64;
        k_pool<<<(total + 255) / 256, 256, 0, stream>>>(buf0, buf1, dpi0, dpw0,
                                                        V1N, V0N, 64, total);
    }
    k_sconv<64, 8, 128, true, false><<<NB * V1N / 16, 128, 0, stream>>>(
        buf1, idx1, ws + W1S, enc_b1, buf0, V1N);
    {
        int total = NB * V2N * 128;
        k_pool<<<(total + 255) / 256, 256, 0, stream>>>(buf0, buf1, dpi1, dpw1,
                                                        V2N, V1N, 128, total);
    }
    k_sconv<128, 16, 256, true, false><<<NB * V2N / 16, 128, 0, stream>>>(
        buf1, idx2, ws + W2S, enc_b2, buf0, V2N);
    {
        int total = NB * V3N * 256;
        k_pool<<<(total + 255) / 256, 256, 0, stream>>>(buf0, buf1, dpi2, dpw2,
                                                        V3N, V2N, 256, total);
    }

    // ---- latent -----------------------------------------------------------
    float* z_out = (float*)d_out + (size_t)NB * V0N * 3;
    {
        dim3 g(16, ZSPLIT);
        k_gemm_z_split<<<g, 32, 0, stream>>>(buf1, ws + WZS, zprt);
        k_z_final<<<(16 * 256 + 255) / 256, 256, 0, stream>>>(zprt, bz, z_out, zbf);
    }
    k_gemm_p<<<5120, 32, 0, stream>>>(zbf, ws + WPS, bp, buf0);

    // ---- decoder ----------------------------------------------------------
    {
        int total = NB * V2N * 256;
        k_pool<<<(total + 255) / 256, 256, 0, stream>>>(buf0, buf1, upi0, upw0,
                                                        V2N, V3N, 256, total);
    }
    k_sconv<256, 8, 128, true, false><<<NB * V2N / 16, 128, 0, stream>>>(
        buf1, idx2, ws + WD0S, dec_b0, buf0, V2N);
    {
        int total = NB * V1N * 128;
        k_pool<<<(total + 255) / 256, 256, 0, stream>>>(buf0, buf1, upi1, upw1,
                                                        V1N, V2N, 128, total);
    }
    k_sconv<128, 4, 64, true, false><<<NB * V1N / 16, 128, 0, stream>>>(
        buf1, idx1, ws + WD1S, dec_b1, buf0, V1N);
    {
        int total = NB * V0N * 64;
        k_pool<<<(total + 255) / 256, 256, 0, stream>>>(buf0, buf1, upi2, upw2,
                                                        V0N, V1N, 64, total);
    }
    // final conv: f32 output straight into d_out (y), no activation
    k_sconv<64, 1, 3, false, true><<<NB * V0N / 16, 128, 0, stream>>>(
        buf1, idx0, ws + WD2S, dec_b2, d_out, V0N);
}
